// ModalityRouter_81853486727572
// MI455X (gfx1250) — matrix operand order repaired and checked by reading, not compiled, not run on hardware
//
#include <hip/hip_runtime.h>
#include <hip/hip_bf16.h>

// ---------------------------------------------------------------------------
// MoE router for MI455X (gfx1250, wave32, WMMA).
//   logits = x[32768,2048] @ gate_w[64,2048]^T   (bf16 WMMA, f32 accumulate)
//   top-2 + softmax + per-expert load
//
// Roofline: x stream = 256 MB @ 23.3 TB/s ~= 11.5 us floor; AI = 32 FLOP/B
// -> needs >745 TFLOPS matmul rate -> bf16 WMMA (16x16x32), not f32 (16x16x4).
// Round-1 lesson: software bf16 packing was VALU-bound; now B is pre-packed
// once into d_ws in WMMA register layout, and A uses v_cvt_pk_bf16_f32.
// ---------------------------------------------------------------------------

typedef __attribute__((ext_vector_type(16))) __bf16 v16bf;
typedef __attribute__((ext_vector_type(8)))  float  v8f;

namespace {
constexpr int D    = 2048;    // hidden dim (K)
constexpr int E    = 64;      // experts (N)
constexpr int NTOK = 32768;   // 4*8192 tokens (M)

constexpr int WAVES          = 4;                      // waves per block
constexpr int TPB            = WAVES * 32;             // 128 threads
constexpr int MT             = 2;                      // 16-token M tiles per wave
constexpr int TOKS_PER_WAVE  = MT * 16;                // 32
constexpr int TOKS_PER_BLOCK = WAVES * TOKS_PER_WAVE;  // 128
constexpr int NT             = E / 16;                 // 4 N tiles
constexpr int KSTEPS         = D / 32;                 // 64 k-blocks

// pre-packed B: per (k-block, n-tile): 32 lanes x 8 DWORDs (16 bf16 per lane)
constexpr int BPK_DWORDS_PER_TILE = 32 * 8;                       // 256
constexpr int BPK_TOTAL           = KSTEPS * NT * BPK_DWORDS_PER_TILE; // 65536 DWORDs = 256 KB

// output layout (flat float buffer)
constexpr int OFF_GATES  = 0;
constexpr int OFF_IDX    = 2 * NTOK;           // 65536
constexpr int OFF_LOAD   = 4 * NTOK;           // 131072
constexpr int OFF_TOPLOG = 4 * NTOK + E;       // 131136
} // namespace

union BF16x16 {
  unsigned u[8];
  uint4    q[2];
  v16bf    v;
};

// hardware RNE f32 pair -> packed bf16x2 (one VALU op)
__device__ __forceinline__ unsigned cvt_pk_bf16(float lo, float hi) {
  unsigned r;
  asm("v_cvt_pk_bf16_f32 %0, %1, %2" : "=v"(r) : "v"(lo), "v"(hi));
  return r;
}

// ---------------------------------------------------------------------------
// Kernel 0: pack gate_w (f32, row-major [E][D]) into WMMA-B register layout.
// B tile (32x16 bf16) for wmma_f32_16x16x32_bf16: lane = hi*16+n16 holds
// expert N = nt*16+n16, K = kb + hi*16 + {0..15} as 8 packed DWORDs.
// ws DWORD index: ((kbIdx*NT + nt) * 32 + lane) * 8 + j
// ---------------------------------------------------------------------------
__global__ __launch_bounds__(256)
void pack_gate_w(const float* __restrict__ gw, unsigned* __restrict__ ws) {
  const int t = blockIdx.x * 256 + threadIdx.x;   // one packed DWORD each
  if (t >= BPK_TOTAL) return;
  const int kbIdx = t >> 10;          // / (NT*32*8)
  const int rem   = t & 1023;
  const int nt    = rem >> 8;
  const int lane  = (rem >> 3) & 31;
  const int j     = rem & 7;
  const int hi    = lane >> 4;
  const int n16   = lane & 15;
  const int expert = nt * 16 + n16;
  const int k      = kbIdx * 32 + hi * 16 + 2 * j;
  const float lo = gw[expert * D + k];
  const float hf = gw[expert * D + k + 1];
  ws[t] = cvt_pk_bf16(lo, hf);
}

// ---------------------------------------------------------------------------
// Kernel 1: fused GEMM + top-2 + softmax
// ---------------------------------------------------------------------------
__global__ __launch_bounds__(TPB)
void moe_router_gemm_topk(const float* __restrict__ x,
                          const unsigned* __restrict__ bw,   // packed gate_w
                          float* __restrict__ out) {
  __shared__ float logits[TOKS_PER_BLOCK * E];   // 32 KB

  const int wave = threadIdx.x >> 5;
  const int lane = threadIdx.x & 31;
  const int hi   = lane >> 4;    // half-wave selector
  const int n16  = lane & 15;    // lane within half

  const int block_tok = blockIdx.x * TOKS_PER_BLOCK;
  const int wave_tok  = block_tok + wave * TOKS_PER_WAVE;

  v8f acc[MT][NT];
#pragma unroll
  for (int mt = 0; mt < MT; ++mt)
#pragma unroll
    for (int nt = 0; nt < NT; ++nt)
      acc[mt][nt] = (v8f)(0.0f);

  // per-lane base pointers
  const float* aprow[MT];
#pragma unroll
  for (int mt = 0; mt < MT; ++mt)
    aprow[mt] = x + (size_t)(wave_tok + mt * 16 + n16) * D + hi * 16;
  const uint4* bbase = (const uint4*)(bw) + lane * 2;  // lane's 8 DWORDs = 2 uint4

  // ---- K loop: bf16 WMMA, K step 32 ----
  for (int kbIdx = 0; kbIdx < KSTEPS; ++kbIdx) {
    const int kb = kbIdx * 32;

    // B tiles: pre-packed, 2 x b128 per tile per lane, fully coalesced
    BF16x16 bt[NT];
#pragma unroll
    for (int nt = 0; nt < NT; ++nt) {
      const uint4* bp = bbase + (size_t)(kbIdx * NT + nt) * 64;  // 256 DWORDs/tile = 64 uint4
      bt[nt].q[0] = bp[0];
      bt[nt].q[1] = bp[1];
    }

#pragma unroll
    for (int mt = 0; mt < MT; ++mt) {
      // A tile: 16x32 bf16. Lane row M = n16; lanes 0-15: K 0-15,
      // lanes 16-31: K 16-31 -> one contiguous 16-float run.
      const float* ap = aprow[mt] + kb;
      float4 a0 = *(const float4*)(ap + 0);
      float4 a1 = *(const float4*)(ap + 4);
      float4 a2 = *(const float4*)(ap + 8);
      float4 a3 = *(const float4*)(ap + 12);
      __builtin_prefetch(ap + 32, 0, 3);         // global_prefetch next k-block
      BF16x16 at;
      at.u[0] = cvt_pk_bf16(a0.x, a0.y);
      at.u[1] = cvt_pk_bf16(a0.z, a0.w);
      at.u[2] = cvt_pk_bf16(a1.x, a1.y);
      at.u[3] = cvt_pk_bf16(a1.z, a1.w);
      at.u[4] = cvt_pk_bf16(a2.x, a2.y);
      at.u[5] = cvt_pk_bf16(a2.z, a2.w);
      at.u[6] = cvt_pk_bf16(a3.x, a3.y);
      at.u[7] = cvt_pk_bf16(a3.z, a3.w);

#pragma unroll
      for (int nt = 0; nt < NT; ++nt) {
        acc[mt][nt] = __builtin_amdgcn_wmma_f32_16x16x32_bf16(
            /*neg_a=*/false, at.v, /*neg_b=*/false, bt[nt].v,
            /*c_mod=*/(short)0, acc[mt][nt],
            /*reuse_a=*/false, /*reuse_b=*/false);
      }
    }
  }

  // ---- scatter logits to LDS (C layout: VGPR r -> row r + 8*hi, N = n16) ----
#pragma unroll
  for (int mt = 0; mt < MT; ++mt)
#pragma unroll
    for (int nt = 0; nt < NT; ++nt)
#pragma unroll
      for (int r = 0; r < 8; ++r) {
        const int tok = wave * TOKS_PER_WAVE + mt * 16 + r + 8 * hi;
        logits[tok * E + nt * 16 + n16] = acc[mt][nt][r];
      }
  __syncthreads();

  // ---- phase 2: one thread per token: top-2 + 2-way softmax ----
  const int tok = threadIdx.x;           // TPB == TOKS_PER_BLOCK
  const float* row = &logits[tok * E];
  float b0v = -3.402823466e38f, b1v = -3.402823466e38f;
  int   b0i = 0, b1i = 0;
  for (int s = 0; s < E; ++s) {
    const int e = (s + tok) & (E - 1);   // rotate start: conflict-free LDS banks
    const float l = row[e];
    if (l > b0v) { b1v = b0v; b1i = b0i; b0v = l; b0i = e; }
    else if (l > b1v) { b1v = l; b1i = e; }
  }
  const float e1  = __expf(b1v - b0v);
  const float inv = 1.0f / (1.0f + e1);

  const int g = block_tok + tok;
  out[OFF_GATES + 2 * g + 0]  = inv;
  out[OFF_GATES + 2 * g + 1]  = e1 * inv;
  out[OFF_IDX + 2 * g + 0]    = (float)b0i;   // indices emitted as floats
  out[OFF_IDX + 2 * g + 1]    = (float)b1i;
  out[OFF_TOPLOG + 2 * g + 0] = b0v;
  out[OFF_TOPLOG + 2 * g + 1] = b1v;
}

// Deterministic per-expert load reduction: one block per expert, fixed-order
// strided accumulation + fixed LDS tree (no atomics -> bit-stable replays).
__global__ __launch_bounds__(256)
void moe_router_load(float* __restrict__ out) {
  __shared__ float red[256];
  const int e = blockIdx.x;
  const float* gates = out + OFF_GATES;
  const float* idxf  = out + OFF_IDX;
  float s = 0.0f;
  for (int i = threadIdx.x; i < 2 * NTOK; i += 256) {
    if ((int)idxf[i] == e) s += gates[i];
  }
  red[threadIdx.x] = s;
  __syncthreads();
  for (int off = 128; off > 0; off >>= 1) {
    if (threadIdx.x < off) red[threadIdx.x] += red[threadIdx.x + off];
    __syncthreads();
  }
  if (threadIdx.x == 0) out[OFF_LOAD + e] = red[0];
}

extern "C" void kernel_launch(void* const* d_in, const int* in_sizes, int n_in,
                              void* d_out, int out_size, void* d_ws, size_t ws_size,
                              hipStream_t stream) {
  const float* x  = (const float*)d_in[0];   // [4,8192,2048] f32
  const float* gw = (const float*)d_in[1];   // [64,2048] f32
  float* out = (float*)d_out;
  unsigned* ws = (unsigned*)d_ws;            // 256 KB packed gate_w

  pack_gate_w<<<BPK_TOTAL / 256, 256, 0, stream>>>(gw, ws);
  const dim3 grid1(NTOK / TOKS_PER_BLOCK);   // 256 blocks
  moe_router_gemm_topk<<<grid1, TPB, 0, stream>>>(x, ws, out);
  moe_router_load<<<E, 256, 0, stream>>>(out);
}
